// ProxyMemoryBank_75582834475211
// MI455X (gfx1250) — matrix operand order repaired and checked by reading, not compiled, not run on hardware
//
#include <hip/hip_runtime.h>
#include <math.h>

typedef __attribute__((ext_vector_type(16))) __bf16 v16bf;
typedef __attribute__((ext_vector_type(8)))  float  v8f;

#define B_DIM 1024
#define D_DIM 2048
#define P_DIM 16384
#define INV_TEMP 20.0f
#define MOM 0.2f

#define MT 32              // B-rows per workgroup (2 WMMA M-tiles)
#define NW 8               // waves per workgroup (wave32)
#define PCOLS 256          // P-columns per tile (8 waves x 2 N-tiles x 16)
#define LDSTR (D_DIM + 8)  // bf16 elems per LDS row (+16B pad: conflict-free)

union BFrag { uint4 q[2]; v16bf v; };

// fp32 -> bf16 (round-to-nearest-even), pack two into a dword
__device__ __forceinline__ unsigned pk2(float a, float b) {
  unsigned ua = __float_as_uint(a), ub = __float_as_uint(b);
  ua += 0x7fffu + ((ua >> 16) & 1u);
  ub += 0x7fffu + ((ub >> 16) & 1u);
  return (ua >> 16) | (ub & 0xffff0000u);
}

__device__ __forceinline__ BFrag load_b_bf16(const unsigned short* p) {
  BFrag b;
  b.q[0] = *(const uint4*)p;        // K 0..7  of this lane's 16-K run
  b.q[1] = *(const uint4*)(p + 8);  // K 8..15
  return b;
}

__device__ __forceinline__ BFrag load_b_f32(const float* p) {
  float4 f0 = *(const float4*)(p);
  float4 f1 = *(const float4*)(p + 4);
  float4 f2 = *(const float4*)(p + 8);
  float4 f3 = *(const float4*)(p + 12);
  BFrag b;
  b.q[0].x = pk2(f0.x, f0.y); b.q[0].y = pk2(f0.z, f0.w);
  b.q[0].z = pk2(f1.x, f1.y); b.q[0].w = pk2(f1.z, f1.w);
  b.q[1].x = pk2(f2.x, f2.y); b.q[1].y = pk2(f2.z, f2.w);
  b.q[1].z = pk2(f3.x, f3.y); b.q[1].w = pk2(f3.z, f3.w);
  return b;
}

// per-lane online logsumexp update over one 16x16 C tile slice
__device__ __forceinline__ void online_update(const v8f& c, float* rm, float* rs) {
#pragma unroll
  for (int v = 0; v < 8; ++v) {
    float cv = c[v] * INV_TEMP;
    float nm = fmaxf(rm[v], cv);
    rs[v] = rs[v] * __expf(rm[v] - nm) + __expf(cv - nm);
    rm[v] = nm;
  }
}

__device__ __forceinline__ void merge_state(float& m, float& s, float mo, float so) {
  float nm = fmaxf(m, mo);
  s = s * __expf(m - nm) + so * __expf(mo - nm);
  m = nm;
}

// ---------------------------------------------------------------------------
// Kernel 0: S fp32 -> bf16 into workspace (read once from HBM/L2)
// ---------------------------------------------------------------------------
__global__ void __launch_bounds__(256) convert_S_kernel(
    const float* __restrict__ S, unsigned short* __restrict__ Sbf) {
  size_t n = (size_t)P_DIM * D_DIM;
  size_t stride = (size_t)gridDim.x * blockDim.x * 4;
  for (size_t i = ((size_t)blockIdx.x * blockDim.x + threadIdx.x) * 4; i < n; i += stride) {
    float4 f = *(const float4*)(S + i);
    *(uint2*)(Sbf + i) = make_uint2(pk2(f.x, f.y), pk2(f.z, f.w));
  }
}

// ---------------------------------------------------------------------------
// Kernel 1: zero loss accumulator, copy storage -> out[1..]
// ---------------------------------------------------------------------------
__global__ void __launch_bounds__(256) init_copy_kernel(
    const float* __restrict__ S, float* __restrict__ out) {
  if (blockIdx.x == 0 && threadIdx.x == 0) out[0] = 0.0f;
  size_t n = (size_t)P_DIM * D_DIM;
  size_t stride = (size_t)gridDim.x * blockDim.x;
  for (size_t i = (size_t)blockIdx.x * blockDim.x + threadIdx.x; i < n; i += stride)
    out[1 + i] = S[i];
}

// ---------------------------------------------------------------------------
// Kernel 2: streaming WMMA GEMM (bf16 MAC, f32 accum) + per-lane online
// logsumexp (no cross-lane traffic in the hot loop).
// One workgroup = 32 rows of X over all P; loss = mean(lse_b - logit_b[label]).
// ---------------------------------------------------------------------------
template <bool PRE>
__global__ void __launch_bounds__(256) loss_kernel(
    const float* __restrict__ X, const float* __restrict__ S,
    const unsigned short* __restrict__ Sbf,
    const int* __restrict__ labels, float* __restrict__ out) {
  extern __shared__ char smem[];
  unsigned short* xs = (unsigned short*)smem;  // 32 x (2048+8) bf16
  float* red_m = (float*)(smem + MT * LDSTR * sizeof(unsigned short));
  float* red_s = red_m + NW * MT;
  float* tval  = red_s + NW * MT;

  const int tid = threadIdx.x;
  const int b0  = blockIdx.x * MT;

  // Stage X tile into LDS as bf16 (row-major, padded rows)
  for (int i = tid * 4; i < MT * D_DIM; i += 256 * 4) {
    int row = i >> 11;            // / D_DIM
    int col = i & (D_DIM - 1);
    const float4 f = *(const float4*)(X + (size_t)(b0 + row) * D_DIM + col);
    *(uint2*)(xs + row * LDSTR + col) = make_uint2(pk2(f.x, f.y), pk2(f.z, f.w));
  }
  __syncthreads();

  const int lane = tid & 31;
  const int wv   = tid >> 5;
  const int nIdx = lane & 15;
  const int half = lane >> 4;

  // running (max,sum) per [mtile*2+ntile][C-slot], purely per-lane
  float rm[4][8], rs[4][8];
#pragma unroll
  for (int t = 0; t < 4; ++t)
#pragma unroll
    for (int v = 0; v < 8; ++v) { rm[t][v] = -3.0e38f; rs[t][v] = 0.f; }

  const unsigned short* a0base = xs + nIdx * LDSTR;         // A rows 0..15
  const unsigned short* a1base = xs + (16 + nIdx) * LDSTR;  // A rows 16..31

  for (int pt = 0; pt < P_DIM; pt += PCOLS) {
    const int p0 = pt + wv * 32 + nIdx;   // N-tile 0 column == S row
    const int p1 = p0 + 16;               // N-tile 1 column
    const unsigned short* sb0 = Sbf + (size_t)p0 * D_DIM;
    const unsigned short* sb1 = Sbf + (size_t)p1 * D_DIM;
    const float* sf0 = S + (size_t)p0 * D_DIM;
    const float* sf1 = S + (size_t)p1 * D_DIM;

    if (pt + PCOLS < P_DIM) {  // prefetch next tile's S rows for this lane
      if (PRE) {
        __builtin_prefetch(sb0 + (size_t)PCOLS * D_DIM, 0, 1);
        __builtin_prefetch(sb1 + (size_t)PCOLS * D_DIM, 0, 1);
      } else {
        __builtin_prefetch(sf0 + (size_t)PCOLS * D_DIM, 0, 1);
        __builtin_prefetch(sf1 + (size_t)PCOLS * D_DIM, 0, 1);
      }
    }

    v8f c00 = {0.f, 0.f, 0.f, 0.f, 0.f, 0.f, 0.f, 0.f};
    v8f c01 = c00, c10 = c00, c11 = c00;

    for (int kc = 0; kc < D_DIM; kc += 32) {
      // B 32x16 fragment: lanes 0-15 K=kc..kc+15, lanes 16-31 K=kc+16..kc+31
      const int kb = kc + half * 16;
      BFrag b0, b1;
      if (PRE) { b0 = load_b_bf16(sb0 + kb); b1 = load_b_bf16(sb1 + kb); }
      else     { b0 = load_b_f32 (sf0 + kb); b1 = load_b_f32 (sf1 + kb); }

      // A 16x32 fragments: elems 0-7 K=ka..ka+7, elems 8-15 K=ka+8..ka+15
      const int ka = kc + half * 16;
      BFrag a0, a1;
      a0.q[0] = *(const uint4*)(a0base + ka);
      a0.q[1] = *(const uint4*)(a0base + ka + 8);
      a1.q[0] = *(const uint4*)(a1base + ka);
      a1.q[1] = *(const uint4*)(a1base + ka + 8);

      c00 = __builtin_amdgcn_wmma_f32_16x16x32_bf16(false, a0.v, false, b0.v, (short)0, c00, false, false);
      c01 = __builtin_amdgcn_wmma_f32_16x16x32_bf16(false, a0.v, false, b1.v, (short)0, c01, false, false);
      c10 = __builtin_amdgcn_wmma_f32_16x16x32_bf16(false, a1.v, false, b0.v, (short)0, c10, false, false);
      c11 = __builtin_amdgcn_wmma_f32_16x16x32_bf16(false, a1.v, false, b1.v, (short)0, c11, false, false);
    }

    online_update(c00, rm[0], rs[0]);
    online_update(c01, rm[1], rs[1]);
    online_update(c10, rm[2], rs[2]);
    online_update(c11, rm[3], rs[3]);
  }

  // Merge N-tile pairs in-register, then 4-step cross-lane merge (per half),
  // then per-wave partials -> LDS.  C layout: lanes 0-15 M=v, lanes 16-31 M=v+8.
#pragma unroll
  for (int t = 0; t < 2; ++t) {
    float* m = rm[2 * t]; float* s = rs[2 * t];
#pragma unroll
    for (int v = 0; v < 8; ++v) {
      merge_state(m[v], s[v], rm[2 * t + 1][v], rs[2 * t + 1][v]);
#pragma unroll
      for (int mask = 1; mask <= 8; mask <<= 1) {
        float mo = __shfl_xor(m[v], mask, 32);
        float so = __shfl_xor(s[v], mask, 32);
        merge_state(m[v], s[v], mo, so);
      }
    }
  }
  if (nIdx == 0) {
#pragma unroll
    for (int v = 0; v < 8; ++v) {
      int m = v + 8 * half;
      red_m[wv * MT + m]      = rm[0][v];
      red_s[wv * MT + m]      = rs[0][v];
      red_m[wv * MT + 16 + m] = rm[2][v];
      red_s[wv * MT + 16 + m] = rs[2][v];
    }
  }

  // Target logits: fp32 dot, 4 rows per wave
#pragma unroll
  for (int i = 0; i < 4; ++i) {
    int r = wv * 4 + i;
    int l = labels[b0 + r];
    const float* xr = X + (size_t)(b0 + r) * D_DIM;
    const float* sr = S + (size_t)l * D_DIM;
    float acc = 0.f;
    for (int k = lane * 4; k < D_DIM; k += 128) {
      float4 xa = *(const float4*)(xr + k);
      float4 sa = *(const float4*)(sr + k);
      acc += xa.x * sa.x + xa.y * sa.y + xa.z * sa.z + xa.w * sa.w;
    }
    acc += __shfl_xor(acc, 1, 32);  acc += __shfl_xor(acc, 2, 32);
    acc += __shfl_xor(acc, 4, 32);  acc += __shfl_xor(acc, 8, 32);
    acc += __shfl_xor(acc, 16, 32);
    if (lane == 0) tval[r] = acc * INV_TEMP;
  }
  __syncthreads();

  // Wave 0 combines the 8 wave-partials per row and accumulates the loss
  if (tid < MT) {
    int r = tid;
    float m = -3.0e38f;
#pragma unroll
    for (int w = 0; w < NW; ++w) m = fmaxf(m, red_m[w * MT + r]);
    float s = 0.f;
#pragma unroll
    for (int w = 0; w < NW; ++w) s += red_s[w * MT + r] * __expf(red_m[w * MT + r] - m);
    float contrib = m + logf(s) - tval[r];
    contrib += __shfl_xor(contrib, 1, 32);  contrib += __shfl_xor(contrib, 2, 32);
    contrib += __shfl_xor(contrib, 4, 32);  contrib += __shfl_xor(contrib, 8, 32);
    contrib += __shfl_xor(contrib, 16, 32);
    if (tid == 0) atomicAdd(out, contrib * (1.0f / (float)B_DIM));
  }
}

// ---------------------------------------------------------------------------
// Kernel 3: sequential EMA + renorm, parallel over label chains.
// ---------------------------------------------------------------------------
__global__ void __launch_bounds__(256) chain_kernel(
    const float* __restrict__ X, const float* __restrict__ S,
    const int* __restrict__ labels, float* __restrict__ outS) {
  __shared__ int isHead;
  __shared__ float wsum[8];
  __shared__ float total;
  const int tid = threadIdx.x;
  const int b = blockIdx.x;
  const int l = labels[b];

  if (tid == 0) isHead = 1;
  __syncthreads();
  for (int i = tid; i < b; i += 256)
    if (labels[i] == l) { isHead = 0; break; }
  __syncthreads();
  if (!isHead) return;

  float r[8];
#pragma unroll
  for (int j = 0; j < 8; ++j) r[j] = S[(size_t)l * D_DIM + j * 256 + tid];

  for (int bb = b; bb < B_DIM; ++bb) {
    if (labels[bb] != l) continue;            // uniform across block
    const float* x = X + (size_t)bb * D_DIM;
    float ss = 0.f;
#pragma unroll
    for (int j = 0; j < 8; ++j) {
      r[j] = MOM * r[j] + (1.0f - MOM) * x[j * 256 + tid];
      ss += r[j] * r[j];
    }
    ss += __shfl_xor(ss, 1, 32);  ss += __shfl_xor(ss, 2, 32);
    ss += __shfl_xor(ss, 4, 32);  ss += __shfl_xor(ss, 8, 32);
    ss += __shfl_xor(ss, 16, 32);
    if ((tid & 31) == 0) wsum[tid >> 5] = ss;
    __syncthreads();
    if (tid == 0) {
      float t = 0.f;
      for (int w = 0; w < 8; ++w) t += wsum[w];
      total = t;
    }
    __syncthreads();
    float inv = 1.0f / sqrtf(total);
#pragma unroll
    for (int j = 0; j < 8; ++j) r[j] *= inv;
  }
#pragma unroll
  for (int j = 0; j < 8; ++j) outS[(size_t)l * D_DIM + j * 256 + tid] = r[j];
}

// ---------------------------------------------------------------------------
extern "C" void kernel_launch(void* const* d_in, const int* in_sizes, int n_in,
                              void* d_out, int out_size, void* d_ws, size_t ws_size,
                              hipStream_t stream) {
  (void)in_sizes; (void)n_in; (void)out_size;
  const float* X      = (const float*)d_in[0];   // input_features (B,D)
  const float* S      = (const float*)d_in[1];   // storage (P,D)
  const int*   absLbl = (const int*)  d_in[4];   // abs_proxy_labels (B,)
  float* out = (float*)d_out;                    // [loss, new_storage...]

  init_copy_kernel<<<2048, 256, 0, stream>>>(S, out);

  size_t shmem = (size_t)MT * LDSTR * sizeof(unsigned short)
               + (size_t)(2 * NW * MT + MT) * sizeof(float);
  const size_t needBf = (size_t)P_DIM * D_DIM * sizeof(unsigned short);

  if (ws_size >= needBf) {
    unsigned short* Sbf = (unsigned short*)d_ws;
    convert_S_kernel<<<2048, 256, 0, stream>>>(S, Sbf);
    hipFuncSetAttribute((const void*)loss_kernel<true>,
                        hipFuncAttributeMaxDynamicSharedMemorySize, (int)shmem);
    loss_kernel<true><<<B_DIM / MT, 256, shmem, stream>>>(X, S, Sbf, absLbl, out);
  } else {
    hipFuncSetAttribute((const void*)loss_kernel<false>,
                        hipFuncAttributeMaxDynamicSharedMemorySize, (int)shmem);
    loss_kernel<false><<<B_DIM / MT, 256, shmem, stream>>>(X, S, nullptr, absLbl, out);
  }
  chain_kernel<<<B_DIM, 256, 0, stream>>>(X, S, absLbl, out + 1);
}
